// TransformerCoupledDecoder_36558761623798
// MI455X (gfx1250) — matrix operand order repaired and checked by reading, not compiled, not run on hardware
//
#include <hip/hip_runtime.h>
#include <hip/hip_bf16.h>

typedef _Float16 half_t;
typedef __attribute__((ext_vector_type(16))) _Float16 v16h;
typedef __attribute__((ext_vector_type(8)))  _Float16 v8h;
typedef __attribute__((ext_vector_type(8)))  float    v8f;

#define LNUM   6
#define DMODEL 256
#define NHEAD  8
#define DHEAD  32
#define DFFN   2048
#define QLEN   300
#define BATCH  64
#define SLEN   1200
#define NOBJ_  81
#define NVERB_ 117
#define WDIM_  300
#define WPAD   320            /* 300 padded to mult-of-32, zero-filled */
#define COGLD  128            /* 117 padded to 128, zero-filled        */
#define NT     (QLEN*BATCH)   /* 19200 tokens, [Q,B] major */
#define NM     (SLEN*BATCH)   /* 76800 memory tokens */

static __device__ __forceinline__ v16h cat8(v8h lo, v8h hi) {
  return __builtin_shufflevector(lo, hi, 0,1,2,3,4,5,6,7,8,9,10,11,12,13,14,15);
}
static __device__ __forceinline__ int imin(int a, int b) { return a < b ? a : b; }

// ---------------------------------------------------------------------------
// WMMA GEMM, 32x32 tile per wave (2x2 subtiles, K stepped by 32).
//   btrans=1: B is [N,K] row-major (A*B^T, i.e. x @ W.T)
//   btrans=0: B is [K,N] row-major (A*B)
// vecok (host: btrans && K%32==0 && lds%8==0): clamped-pointer aligned b128
// loads for ALL tiles (out-of-range rows/cols only pollute outputs that the
// epilogue skips; K is exact so no cross-contamination). Otherwise: scalar
// path with unconditional clamped loads + v_cndmask selects on the K bound.
// Output: f32 (Cf) and/or f16 (Ch); swap=1 scatters [q*B+b] rows to [b*Q+q].
// ---------------------------------------------------------------------------
__global__ __launch_bounds__(256)
void gemm16(const half_t* __restrict__ A, int lda,
            const half_t* __restrict__ Bm, int ldb, int btrans, int vecok,
            const float* __restrict__ bias,
            float* __restrict__ Cf, half_t* __restrict__ Ch,
            int ldc, int M, int N, int K,
            int act, int swap, int Bdim, int Qdim)
{
  const int wid  = threadIdx.x >> 5;
  const int lane = threadIdx.x & 31;
  const int g    = lane >> 4;
  const int ln   = lane & 15;
  const int ntile = (N + 31) >> 5;
  const int mtile = (M + 31) >> 5;
  const int tile  = blockIdx.x * (blockDim.x >> 5) + wid;
  if (tile >= ntile * mtile) return;
  const int mt = tile / ntile, nt = tile % ntile;
  const int m0 = mt * 32, n0 = nt * 32;

  v8f acc00, acc01, acc10, acc11;
  #pragma unroll
  for (int r = 0; r < 8; ++r) { acc00[r]=0.f; acc01[r]=0.f; acc10[r]=0.f; acc11[r]=0.f; }

  // clamped row/col indices: duplicates only feed outputs the epilogue skips
  const int r0 = imin(m0 + ln,      M - 1);
  const int r1 = imin(m0 + 16 + ln, M - 1);
  const int c0 = imin(n0 + ln,      N - 1);
  const int c1 = imin(n0 + 16 + ln, N - 1);
  const half_t* ap0 = A + (long)r0 * lda;
  const half_t* ap1 = A + (long)r1 * lda;

  if (vecok) {
    const half_t* bp0 = Bm + (long)c0 * ldb;
    const half_t* bp1 = Bm + (long)c1 * ldb;
    for (int k0 = 0; k0 < K; k0 += 32) {
      const int ka0 = k0 + g * 16;       // A: 16 contiguous K per lane-half
      const int ka1 = ka0 + 8;
      const int kb0 = k0 + g * 16;       // B: 16 contiguous K per lane-half
      __builtin_prefetch(ap0 + k0 + 64, 0, 0);
      __builtin_prefetch(bp0 + k0 + 64, 0, 0);
      const v16h a0 = cat8(*(const v8h*)(ap0 + ka0), *(const v8h*)(ap0 + ka1));
      const v16h a1 = cat8(*(const v8h*)(ap1 + ka0), *(const v8h*)(ap1 + ka1));
      const v16h b0 = cat8(*(const v8h*)(bp0 + kb0), *(const v8h*)(bp0 + kb0 + 8));
      const v16h b1 = cat8(*(const v8h*)(bp1 + kb0), *(const v8h*)(bp1 + kb0 + 8));
      acc00 = __builtin_amdgcn_wmma_f32_16x16x32_f16(false, a0, false, b0, (short)0, acc00, false, false);
      acc01 = __builtin_amdgcn_wmma_f32_16x16x32_f16(false, a0, false, b1, (short)0, acc01, false, false);
      acc10 = __builtin_amdgcn_wmma_f32_16x16x32_f16(false, a1, false, b0, (short)0, acc10, false, false);
      acc11 = __builtin_amdgcn_wmma_f32_16x16x32_f16(false, a1, false, b1, (short)0, acc11, false, false);
    }
  } else {
    const half_t hz = (half_t)0.0f;
    const half_t* bp0 = Bm + (btrans ? (long)c0 * ldb : (long)c0);
    const half_t* bp1 = Bm + (btrans ? (long)c1 * ldb : (long)c1);
    for (int k0 = 0; k0 < K; k0 += 32) {
      const int ka0 = k0 + g * 16, ka1 = ka0 + 8;
      const int kb0 = k0 + g * 16;
      v16h a0, a1, b0, b1;
      #pragma unroll
      for (int e = 0; e < 8; ++e) {
        const int k1 = ka0 + e, k2 = ka1 + e;
        const int k1c = imin(k1, K - 1);
        const int k2c = imin(k2, K - 1);
        const half_t t00 = ap0[k1c];     // unconditional clamped loads,
        const half_t t01 = ap0[k2c];     // selected below (v_cndmask, no
        const half_t t10 = ap1[k1c];     // exec-mask branches)
        const half_t t11 = ap1[k2c];
        a0[e]     = (k1 < K) ? t00 : hz;
        a0[e + 8] = (k2 < K) ? t01 : hz;
        a1[e]     = (k1 < K) ? t10 : hz;
        a1[e + 8] = (k2 < K) ? t11 : hz;
      }
      #pragma unroll
      for (int e = 0; e < 16; ++e) {
        const int kk  = kb0 + e;
        const int kkc = imin(kk, K - 1);
        const long koff = btrans ? (long)kkc : (long)kkc * ldb;
        const half_t t0 = bp0[koff];
        const half_t t1 = bp1[koff];
        b0[e] = (kk < K) ? t0 : hz;
        b1[e] = (kk < K) ? t1 : hz;
      }
      acc00 = __builtin_amdgcn_wmma_f32_16x16x32_f16(false, a0, false, b0, (short)0, acc00, false, false);
      acc01 = __builtin_amdgcn_wmma_f32_16x16x32_f16(false, a0, false, b1, (short)0, acc01, false, false);
      acc10 = __builtin_amdgcn_wmma_f32_16x16x32_f16(false, a1, false, b0, (short)0, acc10, false, false);
      acc11 = __builtin_amdgcn_wmma_f32_16x16x32_f16(false, a1, false, b1, (short)0, acc11, false, false);
    }
  }

  // epilogue: 2x2 subtiles, bounds-checked stores
  #pragma unroll
  for (int i = 0; i < 2; ++i) {
    #pragma unroll
    for (int j = 0; j < 2; ++j) {
      const v8f* accp = (i == 0) ? (j == 0 ? &acc00 : &acc01)
                                 : (j == 0 ? &acc10 : &acc11);
      const int col = n0 + j * 16 + ln;
      if (col >= N) continue;
      const float bv = bias ? bias[col] : 0.0f;
      #pragma unroll
      for (int r = 0; r < 8; ++r) {
        const int row = m0 + i * 16 + g * 8 + r;
        if (row >= M) continue;
        float v = (*accp)[r] + bv;
        if (act == 1)      v = v > 0.0f ? v : 0.0f;
        else if (act == 2) v = 1.0f / (1.0f + __expf(-v));
        long oidx;
        if (swap) {
          const int q = row / Bdim, bb = row % Bdim;
          oidx = ((long)bb * Qdim + q) * ldc + col;
        } else {
          oidx = (long)row * ldc + col;
        }
        if (Cf) Cf[oidx] = v;
        if (Ch) Ch[oidx] = (half_t)v;
      }
    }
  }
}

// ---------------------------------------------------------------------------
// Fused flash-attention: one wave = 16 query rows of one (b,h); dh = 32.
// Q/K fragments: aligned b128 loads (clamped rows); one WMMA per 16-key slab;
// online softmax; P relayed C-frag -> A-frag through per-wave LDS; two WMMAs
// against V per 32-key block.
// ---------------------------------------------------------------------------
__global__ __launch_bounds__(256)
void attn_fwd(const half_t* __restrict__ Qp, int ldq,
              const half_t* __restrict__ Kp, int ldk,
              const half_t* __restrict__ Vp, int ldv,
              half_t* __restrict__ Op, int ldo,
              int Lq, int Lk, int Bn, float scale)
{
  __shared__ half_t plds[8][16][32];
  const int wid  = threadIdx.x >> 5;
  const int lane = threadIdx.x & 31;
  const int g    = lane >> 4;
  const int ln   = lane & 15;
  const int qtiles = (Lq + 15) >> 4;
  const long wg = (long)blockIdx.x * (blockDim.x >> 5) + wid;
  if (wg >= (long)qtiles * Bn * NHEAD) return;
  const int qt  = (int)(wg / (Bn * NHEAD));
  const int rem = (int)(wg % (Bn * NHEAD));
  const int b   = rem / NHEAD, h = rem % NHEAD;
  const int q0  = qt * 16;

  v16h qf;
  {
    const int rowc = imin(q0 + ln, Lq - 1);
    const half_t* qp = Qp + ((long)rowc * Bn + b) * ldq + h * DHEAD + g * 16;
    qf = cat8(*(const v8h*)qp, *(const v8h*)(qp + 8));
  }

  v8f acc0, acc1, zacc;
  #pragma unroll
  for (int r = 0; r < 8; ++r) { acc0[r] = 0.f; acc1[r] = 0.f; zacc[r] = 0.f; }
  float mrow[8], lrow[8];
  #pragma unroll
  for (int r = 0; r < 8; ++r) { mrow[r] = -1e30f; lrow[r] = 0.f; }

  for (int kb = 0; kb < Lk; kb += 32) {
    v16h kf0, kf1;
    {
      int keyc = imin(kb + ln, Lk - 1);
      const half_t* kp = Kp + ((long)keyc * Bn + b) * ldk + h * DHEAD + g * 16;
      kf0 = cat8(*(const v8h*)kp, *(const v8h*)(kp + 8));
      keyc = imin(kb + 16 + ln, Lk - 1);
      const half_t* kp1 = Kp + ((long)keyc * Bn + b) * ldk + h * DHEAD + g * 16;
      kf1 = cat8(*(const v8h*)kp1, *(const v8h*)(kp1 + 8));
    }
    v8f s0 = __builtin_amdgcn_wmma_f32_16x16x32_f16(false, qf, false, kf0, (short)0, zacc, false, false);
    v8f s1 = __builtin_amdgcn_wmma_f32_16x16x32_f16(false, qf, false, kf1, (short)0, zacc, false, false);
    const bool v0 = (kb + ln) < Lk;
    const bool v1 = (kb + 16 + ln) < Lk;
    #pragma unroll
    for (int r = 0; r < 8; ++r) {
      float a0 = v0 ? s0[r] * scale : -1e30f;
      float a1 = v1 ? s1[r] * scale : -1e30f;
      float mx = fmaxf(a0, a1);
      #pragma unroll
      for (int msk = 8; msk >= 1; msk >>= 1) mx = fmaxf(mx, __shfl_xor(mx, msk, 32));
      const float nm = fmaxf(mrow[r], mx);
      const float p0 = __expf(a0 - nm);
      const float p1 = __expf(a1 - nm);
      float sum = p0 + p1;
      #pragma unroll
      for (int msk = 8; msk >= 1; msk >>= 1) sum += __shfl_xor(sum, msk, 32);
      const float corr = __expf(mrow[r] - nm);
      lrow[r] = lrow[r] * corr + sum;
      mrow[r] = nm;
      acc0[r] *= corr;
      acc1[r] *= corr;
      plds[wid][g * 8 + r][ln]      = (half_t)p0;
      plds[wid][g * 8 + r][16 + ln] = (half_t)p1;
    }
    v16h pf;
    {
      const half_t* pl = &plds[wid][ln][0];
      #pragma unroll
      for (int e = 0; e < 8; ++e) {
        pf[e]     = pl[g * 16 + e];
        pf[e + 8] = pl[g * 16 + 8 + e];
      }
    }
    v16h vf0, vf1;
    #pragma unroll
    for (int e = 0; e < 16; ++e) {
      const int keyc = imin(kb + g * 16 + e, Lk - 1);
      const half_t* vp = Vp + ((long)keyc * Bn + b) * ldv + h * DHEAD;
      vf0[e] = vp[ln];
      vf1[e] = vp[16 + ln];
    }
    acc0 = __builtin_amdgcn_wmma_f32_16x16x32_f16(false, pf, false, vf0, (short)0, acc0, false, false);
    acc1 = __builtin_amdgcn_wmma_f32_16x16x32_f16(false, pf, false, vf1, (short)0, acc1, false, false);
  }

  #pragma unroll
  for (int r = 0; r < 8; ++r) {
    const int row = q0 + g * 8 + r;
    if (row >= Lq) continue;
    const float inv = lrow[r] > 0.f ? 1.f / lrow[r] : 0.f;
    half_t* op = Op + ((long)row * Bn + b) * ldo + h * DHEAD;
    op[ln]      = (half_t)(acc0[r] * inv);
    op[16 + ln] = (half_t)(acc1[r] * inv);
  }
}

// --------------------------- wave-per-row LayerNorm -------------------------
__global__ void ln_k(const float* __restrict__ res, const float* __restrict__ y,
                     const float* __restrict__ sc, const float* __restrict__ bi,
                     float* __restrict__ outf, half_t* __restrict__ outh, int rows)
{
  const int row = blockIdx.x * (blockDim.x >> 5) + (threadIdx.x >> 5);
  if (row >= rows) return;
  const int lane = threadIdx.x & 31;
  const long base = (long)row * DMODEL;
  float v[8], s = 0.f, q = 0.f;
  #pragma unroll
  for (int i = 0; i < 8; ++i) {
    const int c = lane + 32 * i;
    float x = y[base + c];
    if (res) x += res[base + c];
    v[i] = x; s += x; q += x * x;
  }
  #pragma unroll
  for (int m = 16; m >= 1; m >>= 1) { s += __shfl_xor(s, m, 32); q += __shfl_xor(q, m, 32); }
  const float mean = s * (1.f / DMODEL);
  const float var  = q * (1.f / DMODEL) - mean * mean;
  const float rstd = rsqrtf(var + 1e-5f);
  #pragma unroll
  for (int i = 0; i < 8; ++i) {
    const int c = lane + 32 * i;
    const float o = (v[i] - mean) * rstd * sc[c] + bi[c];
    if (outf) outf[base + c] = o;
    if (outh) outh[base + c] = (half_t)o;
  }
}

// out_aug = gate*sem_aug + inter -> hout(f32) and LN(out_aug;vnorm) -> vn16
__global__ void gate_combine(const float* __restrict__ gbuf, const float* __restrict__ semaug,
                             const float* __restrict__ inter,
                             const float* __restrict__ vs, const float* __restrict__ vb,
                             float* __restrict__ hout, half_t* __restrict__ vn, int rows)
{
  const int row = blockIdx.x * (blockDim.x >> 5) + (threadIdx.x >> 5);
  if (row >= rows) return;
  const int lane = threadIdx.x & 31;
  const long base = (long)row * DMODEL;
  float v[8], s = 0.f, q = 0.f;
  #pragma unroll
  for (int i = 0; i < 8; ++i) {
    const int c = lane + 32 * i;
    const float x = gbuf[base + c] * semaug[base + c] + inter[base + c];
    v[i] = x; hout[base + c] = x; s += x; q += x * x;
  }
  #pragma unroll
  for (int m = 16; m >= 1; m >>= 1) { s += __shfl_xor(s, m, 32); q += __shfl_xor(q, m, 32); }
  const float mean = s * (1.f / DMODEL);
  const float var  = q * (1.f / DMODEL) - mean * mean;
  const float rstd = rsqrtf(var + 1e-5f);
  #pragma unroll
  for (int i = 0; i < 8; ++i) {
    const int c = lane + 32 * i;
    vn[base + c] = (half_t)((v[i] - mean) * rstd * vs[c] + vb[c]);
  }
}

// argmax over obj logits ([B,Q,NOBJ] layout) + gather co row -> f16 (ld COGLD)
__global__ void argmax_gather(const float* __restrict__ obj, const float* __restrict__ co,
                              half_t* __restrict__ cog)
{
  const int m = blockIdx.x * blockDim.x + threadIdx.x;  // token = q*B + b
  if (m >= NT) return;
  const int q = m / BATCH, b = m % BATCH;
  const float* p = obj + ((long)b * QLEN + q) * NOBJ_;
  int best = 0; float bv = p[0];
  for (int j = 1; j < NOBJ_; ++j) { const float x = p[j]; if (x > bv) { bv = x; best = j; } }
  const float* cr = co + (long)best * NVERB_;
  half_t* o = cog + (long)m * COGLD;
  for (int j = 0; j < NVERB_; ++j) o[j] = (half_t)cr[j];
  for (int j = NVERB_; j < COGLD; ++j) o[j] = (half_t)0.0f;
}

// ------------------------------ elementwise --------------------------------
__global__ void add2_h(const float* a, const float* b, half_t* o, long n) {
  for (long i = blockIdx.x * (long)blockDim.x + threadIdx.x; i < n; i += (long)gridDim.x * blockDim.x)
    o[i] = (half_t)(a[i] + b[i]);
}
__global__ void cast_h(const float* a, half_t* o, long n) {
  for (long i = blockIdx.x * (long)blockDim.x + threadIdx.x; i < n; i += (long)gridDim.x * blockDim.x)
    o[i] = (half_t)a[i];
}
// cast with row padding: src rows of src_ld -> dst rows of dst_ld (zero pad)
__global__ void cast_pad(const float* a, half_t* o, int src_ld, int dst_ld, long n) {
  for (long i = blockIdx.x * (long)blockDim.x + threadIdx.x; i < n; i += (long)gridDim.x * blockDim.x) {
    const long r = i / dst_ld; const int c = (int)(i % dst_ld);
    o[i] = (half_t)(c < src_ld ? a[r * src_ld + c] : 0.0f);
  }
}
__global__ void copy_f(const float* a, float* o, long n) {
  for (long i = blockIdx.x * (long)blockDim.x + threadIdx.x; i < n; i += (long)gridDim.x * blockDim.x)
    o[i] = a[i];
}
// semanticT[d][v] = relu(H[v][d]) + skip[v][d]; zero-padded columns
__global__ void relu_add_t(const float* __restrict__ Hm, const float* __restrict__ skip,
                           half_t* __restrict__ oT) {
  const long n = (long)DMODEL * COGLD;
  for (long i = blockIdx.x * (long)blockDim.x + threadIdx.x; i < n; i += (long)gridDim.x * blockDim.x) {
    const int d = (int)(i / COGLD), v = (int)(i % COGLD);
    float x = 0.0f;
    if (v < NVERB_) {
      const float t = Hm[(long)v * DMODEL + d];
      x = (t > 0.f ? t : 0.f) + skip[(long)v * DMODEL + d];
    }
    oT[i] = (half_t)x;
  }
}

// ---------------------------------------------------------------------------
extern "C" void kernel_launch(void* const* d_in, const int* in_sizes, int n_in,
                              void* d_out, int out_size, void* d_ws, size_t ws_size,
                              hipStream_t stream)
{
  (void)in_sizes; (void)n_in; (void)out_size; (void)ws_size;
  const float* tgt      = (const float*)d_in[0];
  const float* memory   = (const float*)d_in[1];
  const float* qpos     = (const float*)d_in[2];
  const float* pos      = (const float*)d_in[3];
  const float* sa_in_w  = (const float*)d_in[4];
  const float* sa_in_b  = (const float*)d_in[5];
  const float* sa_out_w = (const float*)d_in[6];
  const float* sa_out_b = (const float*)d_in[7];
  const float* ca_in_w  = (const float*)d_in[8];
  const float* ca_in_b  = (const float*)d_in[9];
  const float* ca_out_w = (const float*)d_in[10];
  const float* ca_out_b = (const float*)d_in[11];
  const float* ln_s     = (const float*)d_in[12];
  const float* ln_b     = (const float*)d_in[13];
  const float* ff1_w    = (const float*)d_in[14];
  const float* ff1_b    = (const float*)d_in[15];
  const float* ff2_w    = (const float*)d_in[16];
  const float* ff2_b    = (const float*)d_in[17];
  const float* sem_w    = (const float*)d_in[18];
  const float* sem_b    = (const float*)d_in[19];
  const float* gate_w   = (const float*)d_in[20];
  const float* gate_b   = (const float*)d_in[21];
  const float* vnorm_s  = (const float*)d_in[22];
  const float* vnorm_b  = (const float*)d_in[23];
  const float* fnorm_s  = (const float*)d_in[24];
  const float* fnorm_b  = (const float*)d_in[25];
  const float* obj_w    = (const float*)d_in[26];
  const float* obj_b    = (const float*)d_in[27];
  const float* verb_w   = (const float*)d_in[28];
  const float* verb_b   = (const float*)d_in[29];
  const float* wemb     = (const float*)d_in[30];
  const float* co       = (const float*)d_in[31];
  float* out = (float*)d_out;

  const long NTD = (long)NT * DMODEL;
  const long NMD = (long)NM * DMODEL;
  const long VERBS_OFF = (long)LNUM * NTD;
  const long OBJS_OFF  = VERBS_OFF + (long)LNUM * BATCH * QLEN * NVERB_;

  char* wsp = (char*)d_ws;
  auto alloc = [&](size_t bytes) -> char* {
    char* r = wsp; wsp += (bytes + 255) & ~(size_t)255; return r;
  };
  auto ah = [&](long n) { return (half_t*)alloc((size_t)n * sizeof(half_t)); };
  auto af = [&](long n) { return (float*)alloc((size_t)n * sizeof(float)); };

  // f16 weights (sem/wemb padded to WPAD columns)
  half_t* w_sa_in  = ah((long)LNUM * 3 * DMODEL * DMODEL);
  half_t* w_sa_out = ah((long)LNUM * DMODEL * DMODEL);
  half_t* w_ca_in  = ah((long)LNUM * 3 * DMODEL * DMODEL);
  half_t* w_ca_out = ah((long)LNUM * DMODEL * DMODEL);
  half_t* w_ff1    = ah((long)LNUM * DFFN * DMODEL);
  half_t* w_ff2    = ah((long)LNUM * DMODEL * DFFN);
  half_t* w_sem    = ah(4L * DMODEL * WPAD);
  half_t* w_gate   = ah((long)DMODEL * DMODEL);
  half_t* w_obj    = ah((long)NOBJ_ * DMODEL);
  half_t* w_verb   = ah((long)NVERB_ * DMODEL);
  half_t* wemb16   = ah((long)NVERB_ * WPAD);
  // f16 activations
  half_t* mempos16 = ah(NMD);
  half_t* mem16    = ah(NMD);
  half_t* qkv16    = ah((long)NT * 3 * DMODEL);   // self QKV / cross Q
  half_t* big16    = ah((long)NM * 2 * DMODEL);   // cross KV, reused as FFN hidden
  half_t* tokA16   = ah(NTD);
  half_t* tokB16   = ah(NTD);
  half_t* inter16  = ah(NTD);
  half_t* vn16     = ah(NTD);
  half_t* cog16    = ah((long)NT * COGLD);
  half_t* sq16     = ah((long)NVERB_ * DMODEL);
  half_t* sk16     = ah((long)NVERB_ * DMODEL);
  half_t* sv16     = ah((long)NVERB_ * DMODEL);
  half_t* G16      = ah((long)NVERB_ * COGLD);
  half_t* semT16   = ah((long)DMODEL * COGLD);    // semantic^T, zero-padded
  // f32 buffers
  float* h_out  = af(NTD);
  float* x1f    = af(NTD);
  float* x2f    = af(NTD);
  float* ybuf   = af(NTD);
  float* semaug = af(NTD);
  float* Hf     = af((long)NVERB_ * DMODEL);
  float* skipf  = af((long)NVERB_ * DMODEL);

  const float scale = 1.0f / sqrtf((float)DHEAD);

  auto ewg = [](long n) { long b = (n + 255) / 256; return (int)(b > 16384 ? 16384 : b); };
  auto gemm = [&](const half_t* A, int lda, const half_t* B, int ldb, int btrans,
                  const float* bias, float* Cf, half_t* Ch, int ldc,
                  int M, int N, int K, int act, int swap) {
    const int vecok = btrans && ((K & 31) == 0) && ((lda & 7) == 0) && ((ldb & 7) == 0);
    const long tiles = (long)((M + 31) / 32) * ((N + 31) / 32);
    const int blocks = (int)((tiles + 7) / 8);
    gemm16<<<blocks, 256, 0, stream>>>(A, lda, B, ldb, btrans, vecok, bias, Cf, Ch,
                                       ldc, M, N, K, act, swap, BATCH, QLEN);
  };
  auto attn = [&](const half_t* Qp, int ldq, const half_t* Kp, int ldk,
                  const half_t* Vp, int ldv, half_t* Op, int ldo, int Lq, int Lk) {
    const long waves = (long)((Lq + 15) / 16) * BATCH * NHEAD;
    attn_fwd<<<(int)((waves + 7) / 8), 256, 0, stream>>>(Qp, ldq, Kp, ldk, Vp, ldv,
                                                         Op, ldo, Lq, Lk, BATCH, scale);
  };
  auto lnrow = [&](const float* res, const float* y, const float* s, const float* b,
                   float* of, half_t* oh) {
    ln_k<<<(NT + 7) / 8, 256, 0, stream>>>(res, y, s, b, of, oh, NT);
  };

  // ---- one-time prep: weight casts, memory casts, semantic graph ----
  cast_h<<<ewg(LNUM*3L*DMODEL*DMODEL),256,0,stream>>>(sa_in_w, w_sa_in, LNUM*3L*DMODEL*DMODEL);
  cast_h<<<ewg(LNUM*(long)DMODEL*DMODEL),256,0,stream>>>(sa_out_w, w_sa_out, LNUM*(long)DMODEL*DMODEL);
  cast_h<<<ewg(LNUM*3L*DMODEL*DMODEL),256,0,stream>>>(ca_in_w, w_ca_in, LNUM*3L*DMODEL*DMODEL);
  cast_h<<<ewg(LNUM*(long)DMODEL*DMODEL),256,0,stream>>>(ca_out_w, w_ca_out, LNUM*(long)DMODEL*DMODEL);
  cast_h<<<ewg(LNUM*(long)DFFN*DMODEL),256,0,stream>>>(ff1_w, w_ff1, LNUM*(long)DFFN*DMODEL);
  cast_h<<<ewg(LNUM*(long)DMODEL*DFFN),256,0,stream>>>(ff2_w, w_ff2, LNUM*(long)DMODEL*DFFN);
  cast_pad<<<ewg(4L*DMODEL*WPAD),256,0,stream>>>(sem_w, w_sem, WDIM_, WPAD, 4L*DMODEL*WPAD);
  cast_h<<<ewg((long)DMODEL*DMODEL),256,0,stream>>>(gate_w, w_gate, (long)DMODEL*DMODEL);
  cast_h<<<ewg((long)NOBJ_*DMODEL),256,0,stream>>>(obj_w, w_obj, (long)NOBJ_*DMODEL);
  cast_h<<<ewg((long)NVERB_*DMODEL),256,0,stream>>>(verb_w, w_verb, (long)NVERB_*DMODEL);
  cast_pad<<<ewg((long)NVERB_*WPAD),256,0,stream>>>(wemb, wemb16, WDIM_, WPAD, (long)NVERB_*WPAD);
  add2_h<<<ewg(NMD),256,0,stream>>>(memory, pos, mempos16, NMD);
  cast_h<<<ewg(NMD),256,0,stream>>>(memory, mem16, NMD);
  copy_f<<<ewg(NTD),256,0,stream>>>(tgt, h_out, NTD);

  // semantic = relu((sq @ sk^T) @ sv) + (wemb @ W3^T + b3)   (K padded to 320)
  gemm(wemb16, WPAD, w_sem + 0L*DMODEL*WPAD, WPAD, 1, sem_b + 0*DMODEL,
       nullptr, sq16, DMODEL, NVERB_, DMODEL, WPAD, 0, 0);
  gemm(wemb16, WPAD, w_sem + 1L*DMODEL*WPAD, WPAD, 1, sem_b + 1*DMODEL,
       nullptr, sk16, DMODEL, NVERB_, DMODEL, WPAD, 0, 0);
  gemm(wemb16, WPAD, w_sem + 2L*DMODEL*WPAD, WPAD, 1, sem_b + 2*DMODEL,
       nullptr, sv16, DMODEL, NVERB_, DMODEL, WPAD, 0, 0);
  gemm(sq16, DMODEL, sk16, DMODEL, 1, nullptr, nullptr, G16, COGLD,
       NVERB_, NVERB_, DMODEL, 0, 0);
  gemm(G16, COGLD, sv16, DMODEL, 0, nullptr, Hf, nullptr, DMODEL,
       NVERB_, DMODEL, NVERB_, 0, 0);
  gemm(wemb16, WPAD, w_sem + 3L*DMODEL*WPAD, WPAD, 1, sem_b + 3*DMODEL,
       skipf, nullptr, DMODEL, NVERB_, DMODEL, WPAD, 0, 0);
  relu_add_t<<<ewg((long)DMODEL*COGLD),256,0,stream>>>(Hf, skipf, semT16);

  // ---- decoder layers ----
  for (int i = 0; i < LNUM; ++i) {
    const half_t* Wsa = w_sa_in + (long)i * 3 * DMODEL * DMODEL;
    const half_t* Wca = w_ca_in + (long)i * 3 * DMODEL * DMODEL;
    const float*  bsa = sa_in_b + (long)i * 3 * DMODEL;
    const float*  bca = ca_in_b + (long)i * 3 * DMODEL;

    // self-attention
    add2_h<<<ewg(NTD),256,0,stream>>>(h_out, qpos, tokA16, NTD);     // q/k input
    cast_h<<<ewg(NTD),256,0,stream>>>(h_out, tokB16, NTD);           // v input
    gemm(tokA16, DMODEL, Wsa, DMODEL, 1, bsa, nullptr, qkv16, 3*DMODEL,
         NT, 2*DMODEL, DMODEL, 0, 0);
    gemm(tokB16, DMODEL, Wsa + 2L*DMODEL*DMODEL, DMODEL, 1, bsa + 2*DMODEL,
         nullptr, qkv16 + 2*DMODEL, 3*DMODEL, NT, DMODEL, DMODEL, 0, 0);
    attn(qkv16, 3*DMODEL, qkv16 + DMODEL, 3*DMODEL, qkv16 + 2*DMODEL, 3*DMODEL,
         tokB16, DMODEL, QLEN, QLEN);
    gemm(tokB16, DMODEL, w_sa_out + (long)i*DMODEL*DMODEL, DMODEL, 1,
         sa_out_b + (long)i*DMODEL, ybuf, nullptr, DMODEL, NT, DMODEL, DMODEL, 0, 0);
    lnrow(h_out, ybuf, ln_s + (i*3+0)*DMODEL, ln_b + (i*3+0)*DMODEL, x1f, nullptr);

    // cross-attention
    add2_h<<<ewg(NTD),256,0,stream>>>(x1f, qpos, tokA16, NTD);
    gemm(tokA16, DMODEL, Wca, DMODEL, 1, bca, nullptr, qkv16, DMODEL,
         NT, DMODEL, DMODEL, 0, 0);
    gemm(mempos16, DMODEL, Wca + 1L*DMODEL*DMODEL, DMODEL, 1, bca + DMODEL,
         nullptr, big16, 2*DMODEL, NM, DMODEL, DMODEL, 0, 0);
    gemm(mem16, DMODEL, Wca + 2L*DMODEL*DMODEL, DMODEL, 1, bca + 2*DMODEL,
         nullptr, big16 + DMODEL, 2*DMODEL, NM, DMODEL, DMODEL, 0, 0);
    attn(qkv16, DMODEL, big16, 2*DMODEL, big16 + DMODEL, 2*DMODEL,
         tokB16, DMODEL, QLEN, SLEN);
    gemm(tokB16, DMODEL, w_ca_out + (long)i*DMODEL*DMODEL, DMODEL, 1,
         ca_out_b + (long)i*DMODEL, ybuf, nullptr, DMODEL, NT, DMODEL, DMODEL, 0, 0);
    lnrow(x1f, ybuf, ln_s + (i*3+1)*DMODEL, ln_b + (i*3+1)*DMODEL, x2f, tokA16);

    // FFN (hidden kept in f16, ReLU fused into GEMM1)
    gemm(tokA16, DMODEL, w_ff1 + (long)i*DFFN*DMODEL, DMODEL, 1,
         ff1_b + (long)i*DFFN, nullptr, big16, DFFN, NT, DFFN, DMODEL, 1, 0);
    gemm(big16, DFFN, w_ff2 + (long)i*DMODEL*DFFN, DFFN, 1,
         ff2_b + (long)i*DMODEL, ybuf, nullptr, DMODEL, NT, DMODEL, DFFN, 0, 0);
    lnrow(x2f, ybuf, ln_s + (i*3+2)*DMODEL, ln_b + (i*3+2)*DMODEL, h_out, nullptr);

    // heads + semantic augmentation
    float* inter_f = out + (long)i * NTD;                      // inters[i]
    lnrow(nullptr, h_out, fnorm_s, fnorm_b, inter_f, inter16);
    gemm(inter16, DMODEL, w_obj, DMODEL, 1, obj_b,
         out + OBJS_OFF + (long)i * BATCH * QLEN * NOBJ_, nullptr, NOBJ_,
         NT, NOBJ_, DMODEL, 0, 1);
    argmax_gather<<<(NT + 255) / 256, 256, 0, stream>>>(
        out + OBJS_OFF + (long)i * BATCH * QLEN * NOBJ_, co, cog16);
    gemm(cog16, COGLD, semT16, COGLD, 1, nullptr, semaug, nullptr, DMODEL,
         NT, DMODEL, COGLD, 0, 0);
    gemm(inter16, DMODEL, w_gate, DMODEL, 1, gate_b, ybuf, nullptr, DMODEL,
         NT, DMODEL, DMODEL, 2, 0);                            // fused sigmoid
    gate_combine<<<(NT + 7) / 8, 256, 0, stream>>>(ybuf, semaug, inter_f,
        vnorm_s, vnorm_b, h_out, vn16, NT);
    gemm(vn16, DMODEL, w_verb, DMODEL, 1, verb_b,
         out + VERBS_OFF + (long)i * BATCH * QLEN * NVERB_, nullptr, NVERB_,
         NT, NVERB_, DMODEL, 0, 1);
  }

  // inters[-1] = LN(final augmented output)
  lnrow(nullptr, h_out, fnorm_s, fnorm_b, out + (long)(LNUM - 1) * NTD, nullptr);
}
